// SelfAttention_23373212025465
// MI455X (gfx1250) — matrix operand order repaired and checked by reading, not compiled, not run on hardware
//
#include <hip/hip_runtime.h>

typedef __attribute__((ext_vector_type(16))) __bf16 v16bf;
typedef __attribute__((ext_vector_type(8)))  __bf16 v8bf;
typedef __attribute__((ext_vector_type(8)))  float  v8f;
typedef __attribute__((ext_vector_type(4)))  unsigned int v4u;
typedef __attribute__((ext_vector_type(8)))  int v8i;
typedef __attribute__((ext_vector_type(4)))  int v4i;

#define DEV static __device__ __forceinline__

constexpr int Bn = 4, T = 2048, D = 1024, H = 16, Hd = 64;
constexpr int BT = Bn * T;     // 8192 rows
constexpr int D3 = 3 * D;      // 3072

// ---- WMMA fragment helpers (bf16, 16x16x32) -------------------------------
// A 16x32 fragment, contiguous k order chosen here (B uses the same): lane m<16 holds row m, K = 0..15; lanes 16..31 K = 16..31.
// rowk0 points at A[row] + k0.
DEV v16bf make_a_frag(const __bf16* rowk0, int hi) {
  return *reinterpret_cast<const v16bf*>(rowk0 + hi * 16);
}
// B 32x16 fragment, same contiguous k order as A: lane n<16 = column n, K = 0..15; lanes 16..31 K = 16..31.
// colk0 points at B^T[col] + k0 (K contiguous in memory or LDS).
DEV v16bf load_b_frag(const __bf16* colk0, int hi) {
  return *reinterpret_cast<const v16bf*>(colk0 + hi * 16);
}
DEV v8f wmma_bf16(v16bf a, v16bf b, v8f c) {
  return __builtin_amdgcn_wmma_f32_16x16x32_bf16(false, a, false, b, (short)0, c, false, false);
}
// Row reductions: C-layout confines each output row to one 16-lane half,
// so xor-shuffles over masks 1,2,4,8 reduce exactly one row.
DEV float half_max(float v) {
  #pragma unroll
  for (int off = 8; off >= 1; off >>= 1) v = fmaxf(v, __shfl_xor(v, off, 32));
  return v;
}
DEV float half_sum(float v) {
  #pragma unroll
  for (int off = 8; off >= 1; off >>= 1) v += __shfl_xor(v, off, 32);
  return v;
}

// ---- Tensor Data Mover descriptor builders (ISA cdna5 §8, D# groups) -------
DEV unsigned lds_addr_of(const void* p) {          // low 32 bits of flat shared ptr = LDS byte addr
  return (unsigned)(size_t)p;
}
DEV v4u tdm_g0(unsigned lds, unsigned long long gaddr) {
  v4u g;
  g.x = 1u;                                        // count=1, user descriptor, no gather
  g.y = lds;                                       // lds_addr [63:32]
  g.z = (unsigned)gaddr;                           // global_addr[31:0]
  g.w = (unsigned)(gaddr >> 32) | (2u << 30);      // global_addr[56:32] | type=2 (image)
  return g;
}
// 2D tile of 2-byte elements: tensor [dim1][dim0] with row stride d0_stride,
// tile [tile_d1][tile_d0] packed into LDS.
DEV v8i tdm_g1(unsigned tensor_d0, unsigned tensor_d1,
               unsigned tile_d0, unsigned tile_d1, unsigned long long d0_stride) {
  unsigned w0 = 1u << 16;                                          // data_size=1 (2 bytes)
  unsigned w1 = (tensor_d0 & 0xFFFFu) << 16;                       // tensor_dim0[15:0] @ bits63:48
  unsigned w2 = ((tensor_d0 >> 16) & 0xFFFFu) | ((tensor_d1 & 0xFFFFu) << 16);
  unsigned w3 = ((tensor_d1 >> 16) & 0xFFFFu) | ((tile_d0 & 0xFFFFu) << 16);
  unsigned w4 = (tile_d1 & 0xFFFFu);                               // tile_dim2 = 0
  unsigned w5 = (unsigned)(d0_stride & 0xFFFFFFFFu);               // dim0_stride[31:0]
  unsigned w6 = (unsigned)((d0_stride >> 32) & 0xFFFFu);           // dim0_stride[47:32]
  unsigned w7 = 0u;                                                // dim1_stride unused (2D)
  v8i g = {(int)w0, (int)w1, (int)w2, (int)w3, (int)w4, (int)w5, (int)w6, (int)w7};
  return g;
}
DEV void tdm_load_2d(unsigned lds, const void* gptr,
                     unsigned td0, unsigned td1, unsigned tl0, unsigned tl1,
                     unsigned long long stride) {
  v4i z4 = {0, 0, 0, 0};
  v8i z8 = {0, 0, 0, 0, 0, 0, 0, 0};
  // 6-arg toolchain variant: (g0, g1, g2, g3, extra, cpol); 2D tile uses only g0/g1.
  __builtin_amdgcn_tensor_load_to_lds(tdm_g0(lds, (unsigned long long)(size_t)gptr),
                                      tdm_g1(td0, td1, tl0, tl1, stride), z4, z4, z8, 0);
}

// ---- conversion / layout kernels ------------------------------------------
__global__ void k_cvt(const float* __restrict__ src, __bf16* __restrict__ dst, int n) {
  for (int i = blockIdx.x * blockDim.x + threadIdx.x; i < n; i += gridDim.x * blockDim.x)
    dst[i] = (__bf16)src[i];
}
// wt[c*K + k] = (bf16) w[k*N + c]   (coalesced writes, gathered reads)
__global__ void k_cvt_t(const float* __restrict__ w, __bf16* __restrict__ wt, int K, int N) {
  int total = K * N;
  for (int i = blockIdx.x * blockDim.x + threadIdx.x; i < total; i += gridDim.x * blockDim.x) {
    int k = i % K, c = i / K;
    wt[i] = (__bf16)w[(size_t)k * N + c];
  }
}

// ---- QKV projection: [8192,1024] x [1024,3072] + bias ----------------------
// Writes Q,K as [bh][T][64] bf16 and V transposed [bh][64][T] bf16.
__global__ void __launch_bounds__(128) k_qkv_gemm(const __bf16* __restrict__ xb,
                                                  const __bf16* __restrict__ wqkvT,
                                                  const float*  __restrict__ bqkv,
                                                  __bf16* __restrict__ qb,
                                                  __bf16* __restrict__ kbuf,
                                                  __bf16* __restrict__ vtb) {
  const int lane = threadIdx.x & 31, wave = threadIdx.x >> 5;
  const int n = lane & 15, hi = lane >> 4;
  const int rt = blockIdx.x;                    // 16-row tile
  const int cg = blockIdx.y * 4 + wave;         // 64-col group (48 total)
  const int c0 = cg * 64;

  const __bf16* arow = xb + (size_t)(rt * 16 + n) * D;
  v8f acc[4] = {};
  for (int k0 = 0; k0 < D; k0 += 32) {
    v16bf a = make_a_frag(arow + k0, hi);
    #pragma unroll
    for (int j = 0; j < 4; ++j)
      acc[j] = wmma_bf16(a, load_b_frag(wqkvT + (size_t)(c0 + j * 16 + n) * D + k0, hi), acc[j]);
  }
  #pragma unroll
  for (int j = 0; j < 4; ++j) {
    const int c = c0 + j * 16 + n;       // uniform matrix/head per 16-col tile
    const float bias = bqkv[c];
    const int cc = c & (D - 1);
    const int h = cc >> 6, d = cc & 63;
    #pragma unroll
    for (int r = 0; r < 8; ++r) {
      const int grow = rt * 16 + r + 8 * hi;          // global row in [0,8192)
      const int b = grow >> 11, t = grow & (T - 1);
      const int bh = b * H + h;
      const float val = acc[j][r] + bias;
      if (c < D)          qb  [((size_t)bh * T + t) * Hd + d] = (__bf16)val;
      else if (c < 2 * D) kbuf[((size_t)bh * T + t) * Hd + d] = (__bf16)val;
      else                vtb [((size_t)bh * Hd + d) * T + t] = (__bf16)val;
    }
  }
}

// ---- causal flash attention ------------------------------------------------
// Block = 4 waves = 4 consecutive 16-row q-tiles of one (b,h).  Wave 0 streams
// the shared 32-key K/V tiles into double-buffered LDS via the Tensor Data
// Mover (TENSORcnt-tracked, one block in flight ahead of compute).
__global__ void __launch_bounds__(128) k_attention(const __bf16* __restrict__ qb,
                                                   const __bf16* __restrict__ kbuf,
                                                   const __bf16* __restrict__ vtb,
                                                   __bf16* __restrict__ attnb) {
  __shared__ __align__(128) __bf16 ksh[2][32 * Hd];   // [key][d]   8 KB
  __shared__ __align__(128) __bf16 vsh[2][Hd * 32];   // [d][key]   8 KB
  __shared__ __align__(128) __bf16 plds[4][16 * 32];  // per-wave P staging 4 KB

  const int lane = threadIdx.x & 31, wave = threadIdx.x >> 5;
  const int n = lane & 15, hi = lane >> 4;

  const int bh = blockIdx.y;
  const int q0 = (blockIdx.x * 4 + wave) * 16;

  const __bf16* qmat = qb   + (size_t)bh * T * Hd;
  const __bf16* kmat = kbuf + (size_t)bh * T * Hd;
  const __bf16* vmat = vtb  + (size_t)bh * Hd * T;

  const __bf16* qrow = qmat + (size_t)(q0 + n) * Hd;
  const v16bf qa0 = make_a_frag(qrow + 0,  hi);   // k-dim 0..31 of Hd
  const v16bf qa1 = make_a_frag(qrow + 32, hi);   // k-dim 32..63

  v8f o[4] = {};
  float rmax[8], rsum[8];
  #pragma unroll
  for (int r = 0; r < 8; ++r) { rmax[r] = -1e30f; rsum[r] = 0.f; }

  const float sc = 0.125f;                        // 1/sqrt(64)
  const int kend = q0 + 15;                       // this wave's causal limit
  const int last_q = (blockIdx.x * 4 + 3) * 16 + 15;
  const int nblk = last_q / 32 + 1;               // shared block count

  // TDM issue: K tile rows=32 keys x 64 d (row stride Hd); V tile rows=64 d x 32 keys (row stride T)
  auto issue = [&](int blk) {
    const int k0 = blk * 32;
    const int buf = blk & 1;
    tdm_load_2d(lds_addr_of(&ksh[buf][0]), kmat + (size_t)k0 * Hd,
                Hd, (unsigned)(T - k0), Hd, 32, Hd);
    tdm_load_2d(lds_addr_of(&vsh[buf][0]), vmat + k0,
                (unsigned)(T - k0), Hd, 32, Hd, T);
  };

  if (wave == 0) issue(0);

  for (int blk = 0; blk < nblk; ++blk) {
    const int k0 = blk * 32;
    const int buf = blk & 1;
    if (wave == 0) {
      if (blk + 1 < nblk) {
        issue(blk + 1);
        __builtin_amdgcn_s_wait_tensorcnt(2);     // block blk's two DMAs retired
      } else {
        __builtin_amdgcn_s_wait_tensorcnt(0);
      }
    }
    __syncthreads();                              // publish LDS tiles to all waves

    if (k0 <= kend) {
      const __bf16* kk = ksh[buf];                // [32][64]
      const __bf16* vv = vsh[buf];                // [64][32]

      // S = Q * K^T for 32 keys (two 16x16 tiles, each 2 WMMAs over Hd), B-frags from LDS
      v8f s0 = {}, s1 = {};
      s0 = wmma_bf16(qa0, load_b_frag(kk + (size_t)n * Hd + 0,  hi), s0);
      s0 = wmma_bf16(qa1, load_b_frag(kk + (size_t)n * Hd + 32, hi), s0);
      s1 = wmma_bf16(qa0, load_b_frag(kk + (size_t)(16 + n) * Hd + 0,  hi), s1);
      s1 = wmma_bf16(qa1, load_b_frag(kk + (size_t)(16 + n) * Hd + 32, hi), s1);

      // mask + scale + online softmax (row stats uniform within each lane-half)
      float p0[8], p1[8], alpha[8];
      #pragma unroll
      for (int r = 0; r < 8; ++r) {
        const int grow = q0 + r + 8 * hi;
        const float v0 = (k0 + n      <= grow) ? s0[r] * sc : -1e30f;
        const float v1 = (k0 + 16 + n <= grow) ? s1[r] * sc : -1e30f;
        const float bm = half_max(fmaxf(v0, v1));
        const float nm = fmaxf(rmax[r], bm);
        alpha[r] = __expf(rmax[r] - nm);
        rmax[r] = nm;
        p0[r] = __expf(v0 - nm);
        p1[r] = __expf(v1 - nm);
        rsum[r] = rsum[r] * alpha[r] + half_sum(p0[r] + p1[r]);
      }
      #pragma unroll
      for (int nt = 0; nt < 4; ++nt)
        #pragma unroll
        for (int r = 0; r < 8; ++r) o[nt][r] *= alpha[r];

      // P: C-layout -> LDS -> A-layout (wave-private slab; DS ops in-order per wave)
      __bf16* pw = plds[wave];
      #pragma unroll
      for (int r = 0; r < 8; ++r) {
        pw[(r + 8 * hi) * 32 + n]      = (__bf16)p0[r];
        pw[(r + 8 * hi) * 32 + n + 16] = (__bf16)p1[r];
      }
      const v16bf pa = make_a_frag(pw + n * 32, hi);

      // O += P[16x32] * V[32x64], V B-frags from LDS
      #pragma unroll
      for (int nt = 0; nt < 4; ++nt)
        o[nt] = wmma_bf16(pa, load_b_frag(vv + (size_t)(nt * 16 + n) * 32, hi), o[nt]);
    }
    __syncthreads();                              // tiles consumed; buffer reusable
  }

  const int b = bh >> 4, h = bh & 15;
  float inv[8];
  #pragma unroll
  for (int r = 0; r < 8; ++r) inv[r] = 1.f / rsum[r];
  #pragma unroll
  for (int nt = 0; nt < 4; ++nt) {
    const int col = h * Hd + nt * 16 + n;
    #pragma unroll
    for (int r = 0; r < 8; ++r) {
      const int t = q0 + r + 8 * hi;
      attnb[((size_t)b * T + t) * D + col] = (__bf16)(o[nt][r] * inv[r]);
    }
  }
}

// ---- output projection: [8192,1024] x [1024,1024] + bias -> fp32 -----------
__global__ void __launch_bounds__(128) k_out_gemm(const __bf16* __restrict__ attnb,
                                                  const __bf16* __restrict__ woutT,
                                                  const float*  __restrict__ bout,
                                                  float* __restrict__ out) {
  const int lane = threadIdx.x & 31, wave = threadIdx.x >> 5;
  const int n = lane & 15, hi = lane >> 4;
  const int rt = blockIdx.x;
  const int c0 = (blockIdx.y * 4 + wave) * 64;

  const __bf16* arow = attnb + (size_t)(rt * 16 + n) * D;
  v8f acc[4] = {};
  for (int k0 = 0; k0 < D; k0 += 32) {
    v16bf a = make_a_frag(arow + k0, hi);
    #pragma unroll
    for (int j = 0; j < 4; ++j)
      acc[j] = wmma_bf16(a, load_b_frag(woutT + (size_t)(c0 + j * 16 + n) * D + k0, hi), acc[j]);
  }
  #pragma unroll
  for (int j = 0; j < 4; ++j) {
    const int c = c0 + j * 16 + n;
    const float bias = bout[c];
    #pragma unroll
    for (int r = 0; r < 8; ++r) {
      const int grow = rt * 16 + r + 8 * hi;
      out[(size_t)grow * D + c] = acc[j][r] + bias;
    }
  }
}

extern "C" void kernel_launch(void* const* d_in, const int* in_sizes, int n_in,
                              void* d_out, int out_size, void* d_ws, size_t ws_size,
                              hipStream_t stream) {
  (void)in_sizes; (void)n_in; (void)out_size; (void)ws_size;
  const float* x    = (const float*)d_in[0];
  const float* Wqkv = (const float*)d_in[1];
  const float* bqkv = (const float*)d_in[2];
  const float* Wout = (const float*)d_in[3];
  const float* bout = (const float*)d_in[4];
  float* out = (float*)d_out;

  char* ws = (char*)d_ws;
  size_t off = 0;
  auto carve = [&](size_t bytes) -> void* {
    void* p = ws + off;
    off += (bytes + 255) & ~(size_t)255;
    return p;
  };
  __bf16* xb    = (__bf16*)carve((size_t)BT * D * 2);        // 16 MB
  __bf16* wqkvT = (__bf16*)carve((size_t)D3 * D * 2);        // 6 MB
  __bf16* woutT = (__bf16*)carve((size_t)D * D * 2);         // 2 MB
  __bf16* qb    = (__bf16*)carve((size_t)Bn * H * T * Hd * 2);
  __bf16* kb    = (__bf16*)carve((size_t)Bn * H * T * Hd * 2);
  __bf16* vtb   = (__bf16*)carve((size_t)Bn * H * Hd * T * 2);
  __bf16* attnb = (__bf16*)carve((size_t)BT * D * 2);        // total ~92 MB

  k_cvt  <<<2048, 256, 0, stream>>>(x, xb, BT * D);
  k_cvt_t<<<2048, 256, 0, stream>>>(Wqkv, wqkvT, D, D3);
  k_cvt_t<<<1024, 256, 0, stream>>>(Wout, woutT, D, D);
  k_qkv_gemm <<<dim3(BT / 16, 12),        128, 0, stream>>>(xb, wqkvT, bqkv, qb, kb, vtb);
  k_attention<<<dim3(T / 16 / 4, Bn * H), 128, 0, stream>>>(qb, kb, vtb, attnb);
  k_out_gemm <<<dim3(BT / 16, 4),         128, 0, stream>>>(attnb, woutT, bout, out);
}
